// SRU_64518998720554
// MI455X (gfx1250) — matrix operand order repaired and checked by reading, not compiled, not run on hardware
//
#include <hip/hip_runtime.h>
#include <hip/hip_bf16.h>

// ---------------- vector types ----------------
typedef __attribute__((ext_vector_type(16))) _Float16     v16h;
typedef __attribute__((ext_vector_type(8)))  float        v8f;
typedef __attribute__((ext_vector_type(4)))  float        v4f;
typedef __attribute__((ext_vector_type(2)))  float        v2f;
typedef __attribute__((ext_vector_type(8)))  unsigned int v8u;
typedef __attribute__((ext_vector_type(4)))  unsigned int v4u;

// ---------------- problem constants ----------------
constexpr int T_STEPS = 512;
constexpr int BATCH   = 256;
constexpr int XD      = 128;
constexpr int RSZ     = 64;
constexpr int PHI     = 256;
constexpr int OUTD    = 128;
constexpr int MU      = 1024;   // PHI * 4 alphas
constexpr int KXR     = XD + RSZ;  // 192

// ---------------- LDS layout (bytes) ----------------
// strides chosen so every ds_load_b128 is 16B aligned and lanes rotate banks
constexpr int LD_WRB = 72;    // f16 row stride for WrB  [1024][72]
constexpr int LD_WXR = 264;   // f16 row stride for WxrB [192][264]
constexpr int LD_MU  = 1028;  // f32 row stride for mu   [16][1028]
constexpr int LD_R16 = 72;    // f16 row stride for r16  [16][72]
constexpr int LD_X16 = 136;   // f16 row stride for x16  [16][136]

constexpr int OFF_WRB  = 0;
constexpr int OFF_WXR  = OFF_WRB  + 1024 * LD_WRB * 2;   // 147456
constexpr int OFF_MU   = OFF_WXR  + 192  * LD_WXR * 2;   // 248832
constexpr int OFF_R16  = OFF_MU   + 16 * LD_MU * 4;      // 314624
constexpr int OFF_X16  = OFF_R16  + 16 * LD_R16 * 2;     // 316928
constexpr int OFF_RED  = OFF_X16  + 16 * LD_X16 * 2;     // 321280
constexpr int OFF_BR   = OFF_RED  + 4 * 256 * 4;         // 325376
constexpr int OFF_BPHI = OFF_BR   + RSZ  * 4;            // 325632
constexpr int OFF_BO   = OFF_BPHI + PHI  * 4;            // 326656
constexpr int SMEM_BYTES = OFF_BO + OUTD * 4;            // 327168 (<= 320KB)
static_assert(SMEM_BYTES <= 320 * 1024, "LDS budget");

// ---------------- helpers ----------------
static __device__ __forceinline__ unsigned int pk2(float lo, float hi) {
  // __builtin_amdgcn_cvt_pkrtz returns __fp16 ext_vector(2); bit-cast to u32
  return __builtin_bit_cast(unsigned int, __builtin_amdgcn_cvt_pkrtz(lo, hi));
}

static __device__ __forceinline__ v8f wmma_f16(v8u a, v8u b, v8f c) {
  return __builtin_amdgcn_wmma_f32_16x16x32_f16(
      false, __builtin_bit_cast(v16h, a),
      false, __builtin_bit_cast(v16h, b),
      (short)0, c, false, false);
}

static __device__ __forceinline__ v8f wmma_f32k4(v2f a, v2f b, v8f c) {
  return __builtin_amdgcn_wmma_f32_16x16x4_f32(
      false, a, false, b, (short)0, c, false, false);
}

// load 16 contiguous f16 (32B) from LDS as v8u
static __device__ __forceinline__ v8u ld16h(const _Float16* p) {
  v4u lo = *(const v4u*)p;
  v4u hi = *(const v4u*)(p + 8);
  v8u r = {lo[0], lo[1], lo[2], lo[3], hi[0], hi[1], hi[2], hi[3]};
  return r;
}

// ---------------- persistent recurrence kernel ----------------
__global__ __launch_bounds__(256) void sru_persistent(
    const float* __restrict__ xin,   // (T, B, XD)
    const float* __restrict__ Wr,    // (RSZ, MU)
    const float* __restrict__ br,    // (RSZ)
    const float* __restrict__ Wxr,   // (PHI, KXR)
    const float* __restrict__ bphi,  // (PHI)
    const float* __restrict__ Wo,    // (OUTD, MU)
    const float* __restrict__ bo,    // (OUTD)
    const float* __restrict__ Amask, // (1, MU)
    float* __restrict__ out)         // (B, OUTD)
{
  extern __shared__ char smem[];
  _Float16* sWrB  = (_Float16*)(smem + OFF_WRB);
  _Float16* sWxrB = (_Float16*)(smem + OFF_WXR);
  float*    sMu   = (float*)   (smem + OFF_MU);
  _Float16* sR16  = (_Float16*)(smem + OFF_R16);
  _Float16* sX16  = (_Float16*)(smem + OFF_X16);
  float*    sRed  = (float*)   (smem + OFF_RED);
  float*    sBr   = (float*)   (smem + OFF_BR);
  float*    sBphi = (float*)   (smem + OFF_BPHI);
  float*    sBo   = (float*)   (smem + OFF_BO);

  const int tid  = threadIdx.x;
  const int lane = tid & 31;
  const int w    = tid >> 5;      // wave id, 8 waves
  const int hs   = lane >> 4;     // lane half
  const int ln   = lane & 15;
  const int b0   = blockIdx.x * 16;

  // ---- one-time: stage weights (f32 -> f16, k-major B layout), biases, zero mu ----
  for (int idx = tid; idx < RSZ * MU; idx += 256) {
    int n = idx >> 10, k = idx & (MU - 1);
    sWrB[k * LD_WRB + n] = (_Float16)Wr[idx];
  }
  for (int idx = tid; idx < PHI * KXR; idx += 256) {
    int n = idx / KXR, k = idx - n * KXR;
    sWxrB[k * LD_WXR + n] = (_Float16)Wxr[idx];
  }
  for (int i = tid; i < RSZ;  i += 256) sBr[i]   = br[i];
  for (int i = tid; i < PHI;  i += 256) sBphi[i] = bphi[i];
  for (int i = tid; i < OUTD; i += 256) sBo[i]   = bo[i];
  for (int i = tid; i < 16 * LD_MU; i += 256) sMu[i] = 0.0f;

  float aA[4]  = {Amask[0], Amask[256], Amask[512], Amask[768]};
  float aB[4]  = {1.0f - aA[0], 1.0f - aA[1], 1.0f - aA[2], 1.0f - aA[3]};

  // x prefetch registers: thread owns row xm, 8 cols starting at xc
  const int xm = tid >> 4;
  const int xc = (tid & 15) * 8;
  v4f xr0, xr1;
  {
    const v4f* xp = (const v4f*)(xin + ((size_t)0 * BATCH + (b0 + xm)) * XD + xc);
    xr0 = xp[0]; xr1 = xp[1];
  }
  __syncthreads();

  // =================== time loop ===================
  for (int t = 0; t < T_STEPS; ++t) {
    // stage x[t] -> LDS (f16), then prefetch x[t+1] into the same regs
    {
      v4u px = {pk2(xr0[0], xr0[1]), pk2(xr0[2], xr0[3]),
                pk2(xr1[0], xr1[1]), pk2(xr1[2], xr1[3])};
      *(v4u*)(sX16 + xm * LD_X16 + xc) = px;
    }
    if (t + 1 < T_STEPS) {
      const v4f* xp = (const v4f*)(xin + ((size_t)(t + 1) * BATCH + (b0 + xm)) * XD + xc);
      xr0 = xp[0]; xr1 = xp[1];
      if (t + 2 < T_STEPS)
        __builtin_prefetch(xin + ((size_t)(t + 2) * BATCH + (b0 + xm)) * XD + xc, 0, 0);
    }

    // ---- GEMM1: r(16x64) = mu(16x1024) @ Wr^T  (4 N-tiles x 2 K-halves) ----
    v8f c1 = {0.f, 0.f, 0.f, 0.f, 0.f, 0.f, 0.f, 0.f};
    {
      const int n0 = (w & 3) * 16;
      const int kb = (w >> 2) * 512;
      const float* arow = sMu + ln * LD_MU;
      #pragma unroll 4
      for (int ks = 0; ks < 16; ++ks) {
        const int k0 = kb + ks * 32;
        const float* ap = arow + k0 + hs * 8;
        v4f a0 = *(const v4f*)(ap);
        v4f a1 = *(const v4f*)(ap + 4);
        v4f a2 = *(const v4f*)(ap + 16);
        v4f a3 = *(const v4f*)(ap + 20);
        v8u au = {pk2(a0[0], a0[1]), pk2(a0[2], a0[3]),
                  pk2(a1[0], a1[1]), pk2(a1[2], a1[3]),
                  pk2(a2[0], a2[1]), pk2(a2[2], a2[3]),
                  pk2(a3[0], a3[1]), pk2(a3[2], a3[3])};
        v8u bu;
        #pragma unroll
        for (int i = 0; i < 8; ++i) {
          const _Float16* q = sWrB + (k0 + hs * 8 + (i & 3) * 2 + (i >> 2) * 16) * LD_WRB + n0 + ln;
          bu[i] = (unsigned int)__builtin_bit_cast(unsigned short, q[0]) | ((unsigned int)__builtin_bit_cast(unsigned short, q[LD_WRB]) << 16);
        }
        c1 = wmma_f16(au, bu, c1);
      }
    }
    // K-split reduction via LDS, then bias+ReLU -> r16 (f16)
    if (w >= 4) {
      float* rp = sRed + (w - 4) * 256 + hs * 128 + ln;
      #pragma unroll
      for (int v = 0; v < 8; ++v) rp[v * 16] = c1[v];
    }
    __syncthreads();
    if (w < 4) {
      const int n0 = w * 16;
      const float* rp = sRed + w * 256 + hs * 128 + ln;
      const float bb = sBr[n0 + ln];
      #pragma unroll
      for (int v = 0; v < 8; ++v) {
        float rv = fmaxf(c1[v] + rp[v * 16] + bb, 0.0f);
        sR16[(v + hs * 8) * LD_R16 + n0 + ln] = (_Float16)rv;
      }
    }
    __syncthreads();  // r16 + x16 ready

    // ---- GEMM2: phi(16x256) = [x|r](16x192) @ Wxr^T  (2 N-tiles per wave) ----
    v8f c2a = {0.f, 0.f, 0.f, 0.f, 0.f, 0.f, 0.f, 0.f};
    v8f c2b = {0.f, 0.f, 0.f, 0.f, 0.f, 0.f, 0.f, 0.f};
    {
      const int n0 = w * 32;
      #pragma unroll
      for (int ks = 0; ks < 6; ++ks) {
        const int k0 = ks * 32;
        const _Float16* ap = (k0 < XD)
            ? (sX16 + ln * LD_X16 + k0 + hs * 8)
            : (sR16 + ln * LD_R16 + (k0 - XD) + hs * 8);
        v4u alo = *(const v4u*)ap;
        v4u ahi = *(const v4u*)(ap + 16);
        v8u au = {alo[0], alo[1], alo[2], alo[3], ahi[0], ahi[1], ahi[2], ahi[3]};
        v8u bua, bub;
        #pragma unroll
        for (int i = 0; i < 8; ++i) {
          const _Float16* q = sWxrB + (k0 + hs * 8 + (i & 3) * 2 + (i >> 2) * 16) * LD_WXR + n0 + ln;
          bua[i] = (unsigned int)__builtin_bit_cast(unsigned short, q[0]) | ((unsigned int)__builtin_bit_cast(unsigned short, q[LD_WXR]) << 16);
          bub[i] = (unsigned int)__builtin_bit_cast(unsigned short, q[16]) | ((unsigned int)__builtin_bit_cast(unsigned short, q[16 + LD_WXR]) << 16);
        }
        c2a = wmma_f16(au, bua, c2a);
        c2b = wmma_f16(au, bub, c2b);
      }
    }

    // ---- fused epilogue: bias + ReLU + 4-rate EMA update of mu (f32, in LDS) ----
    {
      const int n0  = w * 32;
      const int row = hs * 8;
      #pragma unroll
      for (int j = 0; j < 2; ++j) {
        const v8f cc  = j ? c2b : c2a;
        const int col = n0 + j * 16 + ln;
        const float bb = sBphi[col];
        #pragma unroll
        for (int v = 0; v < 8; ++v) {
          float phiv = fmaxf(cc[v] + bb, 0.0f);
          float* mp = sMu + (row + v) * LD_MU + col;
          #pragma unroll
          for (int a = 0; a < 4; ++a) {
            float old = mp[a * 256];
            mp[a * 256] = aA[a] * old + aB[a] * phiv;
          }
        }
      }
    }
    __syncthreads();  // mu ready for next step's GEMM1
  }

  // =================== output layer: out = relu(mu @ Wo^T + bo) ===================
  // exact f32 path: v_wmma_f32_16x16x4_f32, one N-tile (16 cols) per wave
  {
    const int n0 = w * 16;
    v8f c3 = {0.f, 0.f, 0.f, 0.f, 0.f, 0.f, 0.f, 0.f};
    const float* arow = sMu + ln * LD_MU;
    const float* brow = Wo + (size_t)(n0 + ln) * MU;
    #pragma unroll 8
    for (int k0 = 0; k0 < MU; k0 += 4) {
      v2f a2 = *(const v2f*)(arow + k0 + hs * 2);
      v2f b2 = *(const v2f*)(brow + k0 + hs * 2);
      c3 = wmma_f32k4(a2, b2, c3);
    }
    const float bb = sBo[n0 + ln];
    #pragma unroll
    for (int v = 0; v < 8; ++v) {
      float ov = fmaxf(c3[v] + bb, 0.0f);
      out[(size_t)(b0 + hs * 8 + v) * OUTD + n0 + ln] = ov;
    }
  }
}

// ---------------- host launcher ----------------
extern "C" void kernel_launch(void* const* d_in, const int* in_sizes, int n_in,
                              void* d_out, int out_size, void* d_ws, size_t ws_size,
                              hipStream_t stream) {
  (void)in_sizes; (void)n_in; (void)d_ws; (void)ws_size; (void)out_size;
  const float* xin  = (const float*)d_in[0];
  const float* Wr   = (const float*)d_in[1];
  const float* br   = (const float*)d_in[2];
  const float* Wxr  = (const float*)d_in[3];
  const float* bphi = (const float*)d_in[4];
  const float* Wo   = (const float*)d_in[5];
  const float* bo   = (const float*)d_in[6];
  const float* Am   = (const float*)d_in[7];

  (void)hipFuncSetAttribute((const void*)sru_persistent,
                            hipFuncAttributeMaxDynamicSharedMemorySize,
                            SMEM_BYTES);
  sru_persistent<<<BATCH / 16, 256, SMEM_BYTES, stream>>>(
      xin, Wr, br, Wxr, bphi, Wo, bo, Am, (float*)d_out);
}
